// GNNClassifier_23630910063032
// MI455X (gfx1250) — matrix operand order repaired and checked by reading, not compiled, not run on hardware
//
#include <hip/hip_runtime.h>

#define N_NODES 100000
#define N_EDGES 1600000
#define DIN 64
#define DH  64
#define NCLS 16

typedef __attribute__((ext_vector_type(16))) _Float16 v16h;
typedef __attribute__((ext_vector_type(8)))  float    v8f;

// ---------------------------------------------------------------------------
// Zero workspace (agg1, agg2, deg) with 128-bit stores.
// ---------------------------------------------------------------------------
__global__ void zero_ws(float4* __restrict__ p, int n4) {
  int i = blockIdx.x * blockDim.x + threadIdx.x;
  if (i < n4) p[i] = make_float4(0.f, 0.f, 0.f, 0.f);
}

// ---------------------------------------------------------------------------
// Degree count: one thread per edge.
// ---------------------------------------------------------------------------
__global__ void __launch_bounds__(256)
count_deg(const long long* __restrict__ ei, float* __restrict__ deg) {
  int e = blockIdx.x * 256 + threadIdx.x;
  if (e >= N_EDGES) return;
  atomicAdd(&deg[(int)ei[(long long)N_EDGES + e]], 1.0f);
}

// ---------------------------------------------------------------------------
// Edge-parallel push aggregation: 16 threads/edge, float4 gather + 4 f32
// atomics each. x/agg are L2-resident (25.6MB each vs 192MB L2), so the
// random gathers and atomic-adds resolve in L2.
// ---------------------------------------------------------------------------
__global__ void __launch_bounds__(256)
scatter_add(const float* __restrict__ feat, const long long* __restrict__ ei,
            float* __restrict__ agg) {
  int gid = blockIdx.x * 256 + threadIdx.x;          // < E*16 = 25.6M
  int e  = gid >> 4;
  int f4 = (gid & 15) * 4;
  if (e >= N_EDGES) return;
  int s = (int)ei[e];
  int d = (int)ei[(long long)N_EDGES + e];
  float4 v = *(const float4*)&feat[(size_t)s * 64 + f4];
  float* dst = &agg[(size_t)d * 64 + f4];
  atomicAdd(dst + 0, v.x);
  atomicAdd(dst + 1, v.y);
  atomicAdd(dst + 2, v.z);
  atomicAdd(dst + 3, v.w);
}

// ---------------------------------------------------------------------------
// A-fragment builder (CDNA5 16-bit 16x32 A layout, wave32):
// lane L holds row M=L&15; VGPR i holds K pair k = 2i + 8*(L>>4) + (i>=4?8:0).
// Loads merge into global_load_b128 (two 8-float runs per fragment).
// ---------------------------------------------------------------------------
__device__ inline v16h load_a_frag(const float* row, int kb, int half, float scale) {
  v16h a;
#pragma unroll
  for (int i = 0; i < 8; ++i) {
    int k = kb + 2 * i + 8 * half + ((i >= 4) ? 8 : 0);
    a[2 * i]     = (_Float16)(row[k] * scale);
    a[2 * i + 1] = (_Float16)(row[k + 1] * scale);
  }
  return a;
}

// ---------------------------------------------------------------------------
// B-fragment staging: write f16 weights to LDS already in per-lane fragment
// order, so compute-side fetch is one aligned 32B v16h load (2x ds_load_b128)
// instead of 16 ds_load_u16.
// Fragment element (mat,kc,n,lane,j):
//   col = n*16 + (lane&15),  k = kc*32 + j + 8*(lane>>4) + 8*(j>>3)
//   value = W[mat][k*ldw + col]
// ---------------------------------------------------------------------------
__device__ inline void stage_b_frags(_Float16* sFrag, const float* Wl,
                                     const float* Wr, int nTiles, int ldw) {
  int total = 2 * 2 * nTiles * 32 * 16;              // mats * kc * n * lane * j
  for (int idx = threadIdx.x; idx < total; idx += 256) {
    int j    = idx & 15;
    int lane = (idx >> 4) & 31;
    int n    = (idx >> 9) % nTiles;
    int kc   = (idx >> 9) / nTiles;                  // nTiles is 4 or 1
    int mat  = kc >> 1;
    kc &= 1;
    int col = n * 16 + (lane & 15);
    int k   = kc * 32 + j + 8 * (lane >> 4) + 8 * (j >> 3);
    const float* W = (mat == 0) ? Wl : Wr;
    sFrag[idx] = (_Float16)W[k * ldw + col];
  }
}

// fragIndex layout matches stage_b_frags: (((mat*2)+kc)*nTiles + n)
__device__ inline v16h fetch_b_frag(const _Float16* sFrag, int mat, int kc,
                                    int n, int nTiles, int lane) {
  const v16h* fr = (const v16h*)sFrag;
  return fr[(((mat * 2) + kc) * nTiles + n) * 32 + lane];
}

// ---------------------------------------------------------------------------
// Layer 1: h = relu( (agg/deg) @ W_l + x @ W_r + b ), 64 -> 64.
// One wave per 16-row tile; 16 v_wmma_f32_16x16x32_f16 per wave.
// h aliases agg (all loads feed WMMA before any store; tiles own disjoint
// rows), so no __restrict__ on those two.
// ---------------------------------------------------------------------------
__global__ void __launch_bounds__(256)
sage_gemm1(const float* agg, const float* __restrict__ deg,
           const float* __restrict__ x,
           const float* __restrict__ Wl, const float* __restrict__ Wr,
           const float* __restrict__ bias, float* h) {
  __shared__ __align__(32) _Float16 sFrag[2 * 2 * 4 * 32 * 16];   // 16KB
  stage_b_frags(sFrag, Wl, Wr, 4, DH);
  __syncthreads();

  int wave = threadIdx.x >> 5;
  int lane = threadIdx.x & 31;
  int tile = blockIdx.x * 8 + wave;
  if (tile * 16 >= N_NODES) return;
  int base = tile * 16;
  int M = lane & 15;
  int half = lane >> 4;

  const float* aggRow = agg + (size_t)(base + M) * DIN;
  const float* xRow   = x   + (size_t)(base + M) * DIN;
  float rdeg = 1.0f / fmaxf(deg[base + M], 1.0f);

  v8f acc[4];
#pragma unroll
  for (int n = 0; n < 4; ++n) {
    float bv = bias[n * 16 + M];
#pragma unroll
    for (int r = 0; r < 8; ++r) acc[n][r] = bv;
  }

#pragma unroll
  for (int kc = 0; kc < 2; ++kc) {
    int kb = kc * 32;
    v16h am = load_a_frag(aggRow, kb, half, rdeg);   // mean-aggregated input
    v16h ax = load_a_frag(xRow,   kb, half, 1.0f);   // self features
#pragma unroll
    for (int n = 0; n < 4; ++n) {
      v16h bl = fetch_b_frag(sFrag, 0, kc, n, 4, lane);
      v16h br = fetch_b_frag(sFrag, 1, kc, n, 4, lane);
      acc[n] = __builtin_amdgcn_wmma_f32_16x16x32_f16(
          false, am, false, bl, (short)0, acc[n], false, false);
      acc[n] = __builtin_amdgcn_wmma_f32_16x16x32_f16(
          false, ax, false, br, (short)0, acc[n], false, false);
    }
  }

  // C/D layout: VGPR r -> row base + r + 8*half, col = n*16 + M.  ReLU here.
#pragma unroll
  for (int n = 0; n < 4; ++n)
#pragma unroll
    for (int r = 0; r < 8; ++r)
      h[(size_t)(base + r + 8 * half) * DH + n * 16 + M] = fmaxf(acc[n][r], 0.0f);
}

// ---------------------------------------------------------------------------
// Layer 2: out = (agg2/deg) @ W2_l + h @ W2_r + b2, 64 -> 16 (no relu).
// ---------------------------------------------------------------------------
__global__ void __launch_bounds__(256)
sage_gemm2(const float* __restrict__ agg, const float* __restrict__ deg,
           const float* __restrict__ hfeat,
           const float* __restrict__ Wl, const float* __restrict__ Wr,
           const float* __restrict__ bias, float* __restrict__ out) {
  __shared__ __align__(32) _Float16 sFrag[2 * 2 * 1 * 32 * 16];   // 4KB
  stage_b_frags(sFrag, Wl, Wr, 1, NCLS);
  __syncthreads();

  int wave = threadIdx.x >> 5;
  int lane = threadIdx.x & 31;
  int tile = blockIdx.x * 8 + wave;
  if (tile * 16 >= N_NODES) return;
  int base = tile * 16;
  int M = lane & 15;
  int half = lane >> 4;

  const float* aggRow = agg   + (size_t)(base + M) * DH;
  const float* hRow   = hfeat + (size_t)(base + M) * DH;
  float rdeg = 1.0f / fmaxf(deg[base + M], 1.0f);

  v8f acc;
  {
    float bv = bias[M];
#pragma unroll
    for (int r = 0; r < 8; ++r) acc[r] = bv;
  }

#pragma unroll
  for (int kc = 0; kc < 2; ++kc) {
    int kb = kc * 32;
    v16h am = load_a_frag(aggRow, kb, half, rdeg);
    v16h ah = load_a_frag(hRow,   kb, half, 1.0f);
    v16h bl = fetch_b_frag(sFrag, 0, kc, 0, 1, lane);
    v16h br = fetch_b_frag(sFrag, 1, kc, 0, 1, lane);
    acc = __builtin_amdgcn_wmma_f32_16x16x32_f16(
        false, am, false, bl, (short)0, acc, false, false);
    acc = __builtin_amdgcn_wmma_f32_16x16x32_f16(
        false, ah, false, br, (short)0, acc, false, false);
  }

#pragma unroll
  for (int r = 0; r < 8; ++r)
    out[(size_t)(base + r + 8 * half) * NCLS + M] = acc[r];
}

// ---------------------------------------------------------------------------
extern "C" void kernel_launch(void* const* d_in, const int* in_sizes, int n_in,
                              void* d_out, int out_size, void* d_ws, size_t ws_size,
                              hipStream_t stream) {
  const float*     x   = (const float*)d_in[0];
  const long long* ei  = (const long long*)d_in[1];   // int64 [2, E]
  const float*     W1l = (const float*)d_in[2];
  const float*     W1r = (const float*)d_in[3];
  const float*     b1  = (const float*)d_in[4];
  const float*     W2l = (const float*)d_in[5];
  const float*     W2r = (const float*)d_in[6];
  const float*     b2  = (const float*)d_in[7];
  float* out = (float*)d_out;

  // Workspace layout (floats): [agg1|h : N*64][agg2 : N*64][deg : N] = ~51.6MB
  float* ws   = (float*)d_ws;
  float* agg1 = ws;                                   // reused in place as h
  float* agg2 = ws + (size_t)N_NODES * DIN;
  float* deg  = agg2 + (size_t)N_NODES * DH;

  // 1) zero agg1, agg2, deg (12,900,000 floats, divisible by 4)
  size_t nz = (size_t)N_NODES * DIN + (size_t)N_NODES * DH + (size_t)N_NODES;
  int n4 = (int)(nz / 4);
  zero_ws<<<(n4 + 255) / 256, 256, 0, stream>>>((float4*)d_ws, n4);

  // 2) degree + layer-1 aggregation
  count_deg<<<(N_EDGES + 255) / 256, 256, 0, stream>>>(ei, deg);
  int sblocks = (int)(((long long)N_EDGES * 16 + 255) / 256);   // 100000
  scatter_add<<<sblocks, 256, 0, stream>>>(x, ei, agg1);

  // 3) layer-1 GEMM + bias + relu  (h written in place over agg1)
  int tiles = (N_NODES + 15) / 16;        // 6250
  int gblocks = (tiles + 7) / 8;
  sage_gemm1<<<gblocks, 256, 0, stream>>>(agg1, deg, x, W1l, W1r, b1, agg1);

  // 4) layer-2 aggregation (h -> agg2)
  scatter_add<<<sblocks, 256, 0, stream>>>(agg1, ei, agg2);

  // 5) layer-2 GEMM + bias -> out [N, 16]
  sage_gemm2<<<gblocks, 256, 0, stream>>>(agg2, deg, agg1, W2l, W2r, b2, out);
}
